// DRNN_54924041781996
// MI455X (gfx1250) — matrix operand order repaired and checked by reading, not compiled, not run on hardware
//
#include <hip/hip_runtime.h>
#include <hip/hip_bf16.h>
#include <stdint.h>

// ---------------------------------------------------------------------------
// CDNA5 WMMA types / helpers
// ---------------------------------------------------------------------------
typedef __attribute__((ext_vector_type(16))) __bf16 v16bf;
typedef __attribute__((ext_vector_type(2)))  __bf16 v2bf;
typedef __attribute__((ext_vector_type(8)))  float  v8f;

union BFrag { v16bf v; uint32_t u[8]; };

// native bf16 pack (v_cvt_pk_bf16_f32 path, RNE)
__device__ __forceinline__ uint32_t pack2bf(float lo, float hi) {
  v2bf p;
  p.x = (__bf16)lo;
  p.y = (__bf16)hi;
  return __builtin_bit_cast(uint32_t, p);
}
__device__ __forceinline__ float sigm(float x)      { return 1.0f / (1.0f + __expf(-x)); }
__device__ __forceinline__ float tanh_fast(float x) { return 2.0f / (1.0f + __expf(-2.0f * x)) - 1.0f; }

__device__ __forceinline__ v8f wmma_bf16(const BFrag& a, const BFrag& b, v8f c) {
  // D = A(16x32 bf16) * B(32x16 bf16) + C(16x16 f32)
  return __builtin_amdgcn_wmma_f32_16x16x32_bf16(false, a.v, false, b.v,
                                                 (short)0, c, false, false);
}

// A-fragment K index for 16x32 bf16 A (matches B packing): lanes 0-15 hold K 0-15,
// lanes 16-31 hold K 16-31, packed 2 per VGPR.
__device__ __forceinline__ int a_k(int v, int hlf) {
  return hlf * 16 + 2 * v;
}
// XOR swizzle for LDS state buffers (dword granularity), kills 16-way conflicts
__device__ __forceinline__ int lds_dw(int row, int dw) {
  return row * 128 + (dw ^ (row & 15));
}

// ---------------------------------------------------------------------------
// Weight packer: W [N,K] row-major f32 -> bf16 B-fragment tiles.
// Tile (kt,nt): dst[((kt*nttot + ntbase + ntl)*8 + v)*32 + lane] holds
// bf16x2 of W[nt*16 + lane%16, kt*32 + (lane/16)*16 + 2v (,+1)].
// ---------------------------------------------------------------------------
__global__ void pack_b_tiles(const float* __restrict__ src, int ldk,
                             int ktiles, int ntsrc, int ntbase, int nttot,
                             uint32_t* __restrict__ dst) {
  int tid = blockIdx.x * blockDim.x + threadIdx.x;
  int total = ktiles * ntsrc * 256;
  if (tid >= total) return;
  int lane = tid & 31;
  int v    = (tid >> 5) & 7;
  int rest = tid >> 8;
  int ntl  = rest % ntsrc;
  int kt   = rest / ntsrc;
  int n = ntl * 16 + (lane & 15);
  int k = kt * 32 + (lane >> 4) * 16 + 2 * v;
  const float* row = src + (size_t)n * ldk + k;
  dst[(((size_t)(kt * nttot + ntbase + ntl) * 8) + v) * 32 + lane] =
      pack2bf(row[0], row[1]);
}

__global__ void prep_bias(const float* __restrict__ bii, const float* __restrict__ bic,
                          const float* __restrict__ bih, const float* __restrict__ bhh,
                          float* __restrict__ bcomb, float* __restrict__ bhhd) {
  int i = threadIdx.x;
  if (i < 768) bcomb[i] = bii[i] + bic[i] + bih[i];
  else         bhhd[i - 768] = bhh[i - 768];
}

// ---------------------------------------------------------------------------
// Parallel GEMM over all dilated rows: out[r, 0:NT*16] = dx[r,:] @ W^T (+bias)
// r = t*Bd + q ; q = j*32 + b ; original time = t*rate + j  (no padding: 512%r==0)
// One 16x16 output tile per wave.
// ---------------------------------------------------------------------------
template <int KT>
__global__ void gemm_x_kernel(const float* __restrict__ in, int C, int rate,
                              int Bd, const uint32_t* __restrict__ pw, int NT,
                              const float* __restrict__ bias,
                              float* __restrict__ out, int ldo) {
  int wave = threadIdx.x >> 5, lane = threadIdx.x & 31;
  int tile = blockIdx.x * (blockDim.x >> 5) + wave;
  if (tile >= 1024 * NT) return;          // wave-uniform
  int mt = tile / NT, nt = tile % NT;
  int m = lane & 15, hlf = lane >> 4;
  int r = mt * 16 + m;
  int t = r / Bd, q = r - t * Bd;
  int j = q >> 5, b = q & 31;
  int ot = t * rate + j;
  const float* arow = in + ((size_t)ot * 32 + b) * C;
  v8f acc = {0.f, 0.f, 0.f, 0.f, 0.f, 0.f, 0.f, 0.f};
#pragma unroll
  for (int kt = 0; kt < KT; ++kt) {
    BFrag a, w;
#pragma unroll
    for (int v = 0; v < 8; ++v) {
      int k = kt * 32 + a_k(v, hlf);
      a.u[v] = pack2bf(arow[k], arow[k + 1]);
    }
    const uint32_t* bp = pw + ((size_t)(kt * NT + nt) * 8) * 32 + lane;
#pragma unroll
    for (int v = 0; v < 8; ++v) w.u[v] = bp[(size_t)v * 32];
    acc = wmma_bf16(a, w, acc);
  }
  int n = nt * 16 + m;
  float bi = bias ? bias[n] : 0.0f;
#pragma unroll
  for (int p = 0; p < 8; ++p) {
    int row = mt * 16 + p + hlf * 8;
    out[(size_t)row * ldo + n] = acc[p] + bi;
  }
}

// ---------------------------------------------------------------------------
// Persistent recurrent kernel: one workgroup (32 waves), loops over all
// D = 16384/BD time steps. h,c live as bf16 in LDS; f32 cell state in VGPRs.
// Wave owns blocks (mt,nb) strided by 32 => nb = wave&15 is wave-invariant:
// B-fragments are shared across the wave's blocks. Blocks processed in
// chunks of 2 sharing each k-tile's 7 B-fragment loads.
// ---------------------------------------------------------------------------
template <int BD>
__global__ __launch_bounds__(1024) void recurrent_kernel(
    const float* __restrict__ Gx,        // [16384,768] x-proj + combined bias
    const float* __restrict__ skipSrc,   // layer0: Skip [16384,256]; else input [512,32,256]
    int skipDilated,
    const uint32_t* __restrict__ PWh,    // [8kt][64nt] tiles: nt 0-47 = W_ih, 48-63 = W_hh
    const uint32_t* __restrict__ PWic,   // [8kt][48nt] tiles: W_ic
    const float* __restrict__ bhh,       // [256]
    float* __restrict__ outMain,         // [512,32,256]
    float* __restrict__ outTail) {       // [rate,32,256]
  constexpr int RATE  = BD / 32;
  constexpr int D     = 16384 / BD;
  constexpr int NBLK  = (BD / 16) * 16;
  constexpr int BPW   = NBLK / 32;
  constexpr int CHUNK = (BPW >= 2) ? 2 : 1;

  __shared__ __align__(16) uint16_t hbuf[BD * 256];
  __shared__ __align__(16) uint16_t cbuf[BD * 256];

  const int wave = threadIdx.x >> 5, lane = threadIdx.x & 31;
  const int m = lane & 15, hlf = lane >> 4;
  const int nb = wave & 15;              // wave-invariant column block
  const int n  = nb * 16 + m;
  const int mt0 = wave >> 4;             // mt(blk) = mt0 + blk*2

  float cst[BPW][8];
#pragma unroll
  for (int i = 0; i < BPW; ++i)
#pragma unroll
    for (int p = 0; p < 8; ++p) cst[i][p] = 0.0f;

  for (int i = threadIdx.x; i < BD * 128; i += 1024) {
    ((uint32_t*)hbuf)[i] = 0u;
    ((uint32_t*)cbuf)[i] = 0u;
  }
  __syncthreads();

  for (int t = 0; t < D; ++t) {
    if (t + 1 < D)  // warm L2->WGP path for next step's x-projections
      __builtin_prefetch(Gx + (size_t)(t + 1) * BD * 768, 0, 1);

    uint32_t hpk[BPW][4], cpk[BPW][4];
#pragma unroll
    for (int blk0 = 0; blk0 < BPW; blk0 += CHUNK) {
      v8f ai[CHUNK], af[CHUNK], ao[CHUNK], ac[CHUNK];
      // init accumulators from precomputed x-projection / cell bias
      {
        float bh = bhh[n];
#pragma unroll
        for (int c = 0; c < CHUNK; ++c) {
          const int mt = mt0 + (blk0 + c) * 2;
          const float* g0 = Gx + (size_t)(t * BD + mt * 16) * 768;
#pragma unroll
          for (int p = 0; p < 8; ++p) {
            const float* g = g0 + (size_t)(p + hlf * 8) * 768;
            ai[c][p] = g[n];
            af[c][p] = g[256 + n];
            ao[c][p] = g[512 + n];
            ac[c][p] = bh;
          }
        }
      }
#pragma unroll
      for (int kt = 0; kt < 8; ++kt) {
        // 7 shared B-fragments for this k-tile (same nb for all blocks)
        BFrag wi, wf, wo, wc, vi, vf, vo;
        {
          const uint32_t* ph = PWh  + ((size_t)(kt * 64) * 8) * 32 + lane;
          const uint32_t* pc = PWic + ((size_t)(kt * 48) * 8) * 32 + lane;
#pragma unroll
          for (int v = 0; v < 8; ++v) {
            wi.u[v] = ph[((size_t)(nb)      * 8 + v) * 32];
            wf.u[v] = ph[((size_t)(16 + nb) * 8 + v) * 32];
            wo.u[v] = ph[((size_t)(32 + nb) * 8 + v) * 32];
            wc.u[v] = ph[((size_t)(48 + nb) * 8 + v) * 32];
            vi.u[v] = pc[((size_t)(nb)      * 8 + v) * 32];
            vf.u[v] = pc[((size_t)(16 + nb) * 8 + v) * 32];
            vo.u[v] = pc[((size_t)(32 + nb) * 8 + v) * 32];
          }
        }
#pragma unroll
        for (int c = 0; c < CHUNK; ++c) {
          const int mt = mt0 + (blk0 + c) * 2;
          BFrag ha, ca;
          const int arow = mt * 16 + m;
#pragma unroll
          for (int v = 0; v < 8; ++v) {
            int k = kt * 32 + a_k(v, hlf);
            int dw = lds_dw(arow, k >> 1);
            ha.u[v] = ((const uint32_t*)hbuf)[dw];
            ca.u[v] = ((const uint32_t*)cbuf)[dw];
          }
          ai[c] = wmma_bf16(ha, wi, ai[c]); ai[c] = wmma_bf16(ca, vi, ai[c]);
          af[c] = wmma_bf16(ha, wf, af[c]); af[c] = wmma_bf16(ca, vf, af[c]);
          ao[c] = wmma_bf16(ha, wo, ao[c]); ao[c] = wmma_bf16(ca, vo, ao[c]);
          ac[c] = wmma_bf16(ha, wc, ac[c]);
        }
      }
      // elementwise LSTM update for the chunk's 16x16 blocks
#pragma unroll
      for (int c = 0; c < CHUNK; ++c) {
        const int blk = blk0 + c;
        const int mt = mt0 + blk * 2;
        float hy8[8], cy8[8];
#pragma unroll
        for (int p = 0; p < 8; ++p) {
          float ig = sigm(ai[c][p]);
          float fg = sigm(af[c][p]);
          float og = sigm(ao[c][p]);
          float cg = tanh_fast(ac[c][p]);
          float cy = fg * cst[blk][p] + ig * cg;
          float ry = tanh_fast(cy);
          const int row = mt * 16 + p + hlf * 8;
          float sk;
          if (skipDilated) {
            sk = skipSrc[(size_t)(t * BD + row) * 256 + n];
          } else {
            int jj = row >> 5, bb = row & 31;
            int ot = t * RATE + jj;
            sk = skipSrc[((size_t)ot * 32 + bb) * 256 + n];
          }
          float hy = og * (ry + sk);
          cst[blk][p] = cy;
          hy8[p] = hy; cy8[p] = cy;
          int jj2 = row >> 5, bb2 = row & 31;
          int ot2 = t * RATE + jj2;
          outMain[((size_t)ot2 * 32 + bb2) * 256 + n] = hy;
          int tt = ot2 - (512 - RATE);
          if (tt >= 0) outTail[((size_t)tt * 32 + bb2) * 256 + n] = hy;
        }
#pragma unroll
        for (int a = 0; a < 4; ++a) {
          hpk[blk][a] = pack2bf(hy8[2 * a], hy8[2 * a + 1]);
          cpk[blk][a] = pack2bf(cy8[2 * a], cy8[2 * a + 1]);
        }
      }
    }
    __syncthreads();  // all reads of old h/c complete
#pragma unroll
    for (int blk = 0; blk < BPW; ++blk) {
      const int mt = mt0 + blk * 2;
#pragma unroll
      for (int a = 0; a < 4; ++a) {
        int r0 = mt * 16 + 2 * a + hlf * 8;
        int r1 = r0 + 1;
        hbuf[lds_dw(r0, n >> 1) * 2 + (n & 1)] = (uint16_t)(hpk[blk][a] & 0xFFFFu);
        hbuf[lds_dw(r1, n >> 1) * 2 + (n & 1)] = (uint16_t)(hpk[blk][a] >> 16);
        cbuf[lds_dw(r0, n >> 1) * 2 + (n & 1)] = (uint16_t)(cpk[blk][a] & 0xFFFFu);
        cbuf[lds_dw(r1, n >> 1) * 2 + (n & 1)] = (uint16_t)(cpk[blk][a] >> 16);
      }
    }
    __syncthreads();  // new state visible before next step
  }
}

// ---------------------------------------------------------------------------
// Host launch
// ---------------------------------------------------------------------------
extern "C" void kernel_launch(void* const* d_in, const int* in_sizes, int n_in,
                              void* d_out, int out_size, void* d_ws, size_t ws_size,
                              hipStream_t stream) {
  (void)in_sizes; (void)n_in; (void)out_size; (void)ws_size;
  const float* x = (const float*)d_in[0];
  auto P = [&](int l, int j) { return (const float*)d_in[1 + l * 9 + j]; };
  // dict order: 0 W_ii, 1 W_ic, 2 W_ih, 3 b_ii, 4 b_ic, 5 b_ih, 6 W_hh, 7 b_hh, 8 W_ir

  uint8_t* ws = (uint8_t*)d_ws;
  size_t off = 0;
  auto alloc = [&](size_t bytes) -> void* {
    void* p = ws + off;
    off = (off + bytes + 255) & ~(size_t)255;
    return p;
  };
  float* bufO0 = (float*)alloc((size_t)16384 * 256 * 4);
  float* bufO1 = (float*)alloc((size_t)16384 * 256 * 4);
  float* Gx    = (float*)alloc((size_t)16384 * 768 * 4);
  float* Skip0 = (float*)alloc((size_t)16384 * 256 * 4);
  uint32_t *PWh[4], *PWic[4], *PWii[4], *PWir0;
  float *bcomb[4], *bhhd[4];
  for (int l = 0; l < 4; ++l) {
    PWh[l]  = (uint32_t*)alloc((size_t)8 * 64 * 256 * 4);
    PWic[l] = (uint32_t*)alloc((size_t)8 * 48 * 256 * 4);
    int ktii = (l == 0) ? 2 : 8;
    PWii[l] = (uint32_t*)alloc((size_t)ktii * 48 * 256 * 4);
    bcomb[l] = (float*)alloc(768 * 4);
    bhhd[l]  = (float*)alloc(256 * 4);
  }
  PWir0 = (uint32_t*)alloc((size_t)2 * 16 * 256 * 4);

  // ---- pack weights (parallel, once per call) ----
  for (int l = 0; l < 4; ++l) {
    int C = (l == 0) ? 64 : 256;
    int ktC = C / 32;
    { int tot = 8 * 48 * 256;   // W_ih -> PWh nt 0..47
      pack_b_tiles<<<(tot + 255) / 256, 256, 0, stream>>>(P(l, 2), 256, 8, 48, 0, 64, PWh[l]); }
    { int tot = 8 * 16 * 256;   // W_hh -> PWh nt 48..63
      pack_b_tiles<<<(tot + 255) / 256, 256, 0, stream>>>(P(l, 6), 256, 8, 16, 48, 64, PWh[l]); }
    { int tot = 8 * 48 * 256;   // W_ic
      pack_b_tiles<<<(tot + 255) / 256, 256, 0, stream>>>(P(l, 1), 256, 8, 48, 0, 48, PWic[l]); }
    { int tot = ktC * 48 * 256; // W_ii
      pack_b_tiles<<<(tot + 255) / 256, 256, 0, stream>>>(P(l, 0), C, ktC, 48, 0, 48, PWii[l]); }
    prep_bias<<<1, 1024, 0, stream>>>(P(l, 3), P(l, 4), P(l, 5), P(l, 7), bcomb[l], bhhd[l]);
  }
  { int tot = 2 * 16 * 256;     // W_ir (layer 0 only, [256,64])
    pack_b_tiles<<<(tot + 255) / 256, 256, 0, stream>>>(P(0, 8), 64, 2, 16, 0, 16, PWir0); }

  float* outp = (float*)d_out;
  float* tailbase = outp + (size_t)512 * 32 * 256;
  float* tails[4] = {tailbase, tailbase + 8192, tailbase + 8192 * 3, tailbase + 8192 * 7};
  float* outs[4]  = {bufO0, bufO1, bufO0, outp};

  const float* inbuf = x;
  for (int l = 0; l < 4; ++l) {
    int rate = 1 << l;
    int Bd = 32 * rate;
    int C = (l == 0) ? 64 : 256;
    // hoisted x-projection: Gx = dx @ W_ii^T + (b_ii+b_ic+b_ih)
    if (l == 0)
      gemm_x_kernel<2><<<(1024 * 48) / 8, 256, 0, stream>>>(inbuf, C, rate, Bd, PWii[l], 48, bcomb[l], Gx, 768);
    else
      gemm_x_kernel<8><<<(1024 * 48) / 8, 256, 0, stream>>>(inbuf, C, rate, Bd, PWii[l], 48, bcomb[l], Gx, 768);
    const float* skip; int skipDil;
    if (l == 0) {
      gemm_x_kernel<2><<<(1024 * 16) / 8, 256, 0, stream>>>(inbuf, C, rate, Bd, PWir0, 16,
                                                            (const float*)nullptr, Skip0, 256);
      skip = Skip0; skipDil = 1;
    } else { skip = inbuf; skipDil = 0; }

    switch (Bd) {
      case 32:  recurrent_kernel<32> <<<1, 1024, 0, stream>>>(Gx, skip, skipDil, PWh[l], PWic[l], bhhd[l], outs[l], tails[l]); break;
      case 64:  recurrent_kernel<64> <<<1, 1024, 0, stream>>>(Gx, skip, skipDil, PWh[l], PWic[l], bhhd[l], outs[l], tails[l]); break;
      case 128: recurrent_kernel<128><<<1, 1024, 0, stream>>>(Gx, skip, skipDil, PWh[l], PWic[l], bhhd[l], outs[l], tails[l]); break;
      default:  recurrent_kernel<256><<<1, 1024, 0, stream>>>(Gx, skip, skipDil, PWh[l], PWic[l], bhhd[l], outs[l], tails[l]); break;
    }
    inbuf = outs[l];
  }
}
